// SharedMemoryBank_39676907888656
// MI455X (gfx1250) — matrix operand order repaired and checked by reading, not compiled, not run on hardware
//
#include <hip/hip_runtime.h>
#include <hip/hip_bf16.h>
#include <math.h>

// Problem constants (fixed by the reference)
#define N_TOK 16384
#define DIM   2048
#define KSLOT 8
#define HEADS 8
#define HDIM  256
#define LN_EPS 1e-5f

typedef __bf16 bf16;
typedef __attribute__((ext_vector_type(16))) __bf16 v16bf;
typedef __attribute__((ext_vector_type(8)))  __bf16 v8bf;   // 16-byte chunk
typedef __attribute__((ext_vector_type(8)))  float  v8f;
typedef int v4i_vec __attribute__((vector_size(16)));        // matches builtin param type

// gfx1250 async global->LDS copy (ASYNCcnt-tracked). Guarded: fall back to a
// direct copy through VGPRs if this toolchain doesn't expose the builtin.
#if defined(__AMDGCN__) && __has_builtin(__builtin_amdgcn_global_load_async_to_lds_b128)
#define HAVE_ASYNC_LDS 1
#else
#define HAVE_ASYNC_LDS 0
#endif

// ---------------- wave32 reductions ----------------
__device__ __forceinline__ float wave_sum(float v) {
#pragma unroll
  for (int off = 16; off > 0; off >>= 1) v += __shfl_xor(v, off, 32);
  return v;
}
__device__ __forceinline__ float wave_max(float v) {
#pragma unroll
  for (int off = 16; off > 0; off >>= 1) v = fmaxf(v, __shfl_xor(v, off, 32));
  return v;
}

// ---------------- f32 -> bf16 convert ----------------
__global__ void cvt_f32_bf16(const float* __restrict__ src, bf16* __restrict__ dst, int n) {
  int i = blockIdx.x * blockDim.x + threadIdx.x;
  int stride = gridDim.x * blockDim.x;
  for (; i < n; i += stride) dst[i] = (bf16)src[i];
}

// ---------------- tiny GEMVs: k,v,wq = memory @ W.T  ([8,2048]) ----------------
__global__ void small_rowmats(const float* __restrict__ mem,
                              const float* __restrict__ Wk, const float* __restrict__ Wv,
                              const float* __restrict__ Wq,
                              float* __restrict__ kf, float* __restrict__ vf,
                              float* __restrict__ qf) {
  const int mat = blockIdx.x >> 3;
  const int kk  = blockIdx.x & 7;
  const float* W = (mat == 0) ? Wk : (mat == 1) ? Wv : Wq;
  float* out = (mat == 0) ? kf : (mat == 1) ? vf : qf;
  __shared__ float row[DIM];
  for (int d = threadIdx.x; d < DIM; d += blockDim.x) row[d] = mem[kk * DIM + d];
  __syncthreads();
  for (int c = threadIdx.x; c < DIM; c += blockDim.x) {
    const float* wr = W + (size_t)c * DIM;
    float acc = 0.f;
    for (int d = 0; d < DIM; ++d) acc = fmaf(row[d], wr[d], acc);
    out[kk * DIM + c] = acc;
  }
}

// Stage one 128x32 bf16 tile (rows rowBase..rowBase+127, cols k0..k0+31) into LDS.
// 512 16B chunks, 256 threads -> 2 chunks each.
__device__ __forceinline__ void stage_tile(const bf16* __restrict__ G, int rowBase, int k0,
                                           bf16* S, int tid) {
#pragma unroll
  for (int i = 0; i < 2; ++i) {
    const int c = tid + i * 256;
    const int row = c >> 2, seg = c & 3;
    const bf16* g = G + (size_t)(rowBase + row) * DIM + k0 + seg * 8;
    bf16* l = S + row * 32 + seg * 8;
#if HAVE_ASYNC_LDS
    // builtin expects (global v4i*, lds v4i*, imm offset, imm cpol);
    // generic LDS pointers carry the LDS byte offset in addr[31:0] (aperture rule)
    __builtin_amdgcn_global_load_async_to_lds_b128(
        (__attribute__((address_space(1))) v4i_vec*)(unsigned long long)(uintptr_t)g,
        (__attribute__((address_space(3))) v4i_vec*)(unsigned)(uintptr_t)l,
        0, 0);
#else
    *(v8bf*)l = *(const v8bf*)g;
#endif
  }
}

__device__ __forceinline__ void wait_stage() {
#if HAVE_ASYNC_LDS
  asm volatile("s_wait_asynccnt 0" ::: "memory");
#endif
  __syncthreads();
}

// ---------------- WMMA GEMM: Y[N,2048] = X[N,2048](bf16) @ W[2048,2048]^T(bf16) ----------------
// LDS double-buffered 128x128 workgroup tile; 8 waves, each 32x64 (2x4 WMMA).
// mode 0: f32 out      mode 1: bf16 out      mode 2: f32 out = residual + sigmoid(gate)*Y
__global__ __launch_bounds__(256) void gemm_bf16_nt(
    const bf16* __restrict__ X, const bf16* __restrict__ W,
    float* __restrict__ outF, bf16* __restrict__ outB,
    const float* __restrict__ residual, const float* __restrict__ gate, int mode) {
  __shared__ __align__(16) bf16 sA[2][128 * 32];
  __shared__ __align__(16) bf16 sB[2][128 * 32];

  const int tid   = threadIdx.x;
  const int lane  = tid & 31;
  const int wave  = tid >> 5;
  const int waveM = wave & 3, waveN = wave >> 2;   // 4x2 wave grid
  const int rowBase = blockIdx.y * 128;            // token rows
  const int colBase = blockIdx.x * 128;            // output cols
  const int rA = lane & 15, hA = lane >> 4;

  v8f acc[2][4] = {};

  stage_tile(X, rowBase, 0, sA[0], tid);
  stage_tile(W, colBase, 0, sB[0], tid);
  wait_stage();

  int buf = 0;
  for (int k0 = 0; k0 < DIM; k0 += 32) {
    if (k0 + 32 < DIM) {    // overlap next-stage copy with WMMA on current buffer
      stage_tile(X, rowBase, k0 + 32, sA[buf ^ 1], tid);
      stage_tile(W, colBase, k0 + 32, sB[buf ^ 1], tid);
    }
    if (k0 + 64 < DIM) {    // warm L2 one stage further ahead (global_prefetch_b8)
      __builtin_prefetch(X + (size_t)(rowBase + (tid >> 2)) * DIM + k0 + 64 + (tid & 3) * 8, 0, 1);
      __builtin_prefetch(W + (size_t)(colBase + (tid >> 2)) * DIM + k0 + 64 + (tid & 3) * 8, 0, 1);
    }

    v16bf a[2], b[4];
#pragma unroll
    for (int mi = 0; mi < 2; ++mi) {
      // A 16x32 fragment: lane-half splits each row's 32 K-values into 16B chunks
      const bf16* p = &sA[buf][(waveM * 32 + mi * 16 + rA) * 32 + hA * 8];
      ((v8bf*)&a[mi])[0] = *(const v8bf*)p;
      ((v8bf*)&a[mi])[1] = *(const v8bf*)(p + 16);
    }
#pragma unroll
    for (int ni = 0; ni < 4; ++ni) {
      // B 32x16 fragment: lane = output column (row of W), K split across lane halves
      const bf16* p = &sB[buf][(waveN * 64 + ni * 16 + rA) * 32 + hA * 8];
      ((v8bf*)&b[ni])[0] = *(const v8bf*)p;
      ((v8bf*)&b[ni])[1] = *(const v8bf*)(p + 16);
    }
#pragma unroll
    for (int mi = 0; mi < 2; ++mi)
#pragma unroll
      for (int ni = 0; ni < 4; ++ni)
        acc[mi][ni] = __builtin_amdgcn_wmma_f32_16x16x32_bf16(
            false, a[mi], false, b[ni], (short)0, acc[mi][ni], false, false);

    wait_stage();
    buf ^= 1;
  }

  float g = 0.f;
  if (mode == 2) g = 1.f / (1.f + __expf(-gate[0]));

  const int row0 = rowBase + waveM * 32;
  const int col0 = colBase + waveN * 64;
#pragma unroll
  for (int mi = 0; mi < 2; ++mi) {
#pragma unroll
    for (int ni = 0; ni < 4; ++ni) {
      const int n = col0 + ni * 16 + rA;
#pragma unroll
      for (int j = 0; j < 8; ++j) {
        const int m = row0 + mi * 16 + hA * 8 + j;   // C/D layout: M = vgpr + 8*(lane>=16)
        const size_t idx = (size_t)m * DIM + n;
        const float v = acc[mi][ni][j];
        if (mode == 0)      outF[idx] = v;
        else if (mode == 1) outB[idx] = (bf16)v;
        else                outF[idx] = residual[idx] + g * v;
      }
    }
  }
}

// ---------------- fused read attention (K=8 slots) + LayerNorm, one token per block ----------------
__global__ __launch_bounds__(256) void read_attn_ln(
    const bf16* __restrict__ qb, const float* __restrict__ kf, const float* __restrict__ vf,
    const float* __restrict__ gamma, const float* __restrict__ beta, bf16* __restrict__ lnOut) {
  const int n = blockIdx.x;
  const int tid = threadIdx.x;
  const int lane = tid & 31;
  const int wave = tid >> 5;       // wave == head (32 lanes * 8 dims = 256 = HDIM)
  const int d0 = tid * 8;

  float qv[8];
#pragma unroll
  for (int j = 0; j < 8; ++j) qv[j] = (float)qb[(size_t)n * DIM + d0 + j];

  float s[KSLOT];
#pragma unroll
  for (int kk = 0; kk < KSLOT; ++kk) {
    const float* kr = kf + kk * DIM + d0;
    float acc = 0.f;
#pragma unroll
    for (int j = 0; j < 8; ++j) acc = fmaf(qv[j], kr[j], acc);
    s[kk] = acc;
  }
#pragma unroll
  for (int kk = 0; kk < KSLOT; ++kk) s[kk] = wave_sum(s[kk]);   // per-head dot

  float mx = -3.402823466e38f;
#pragma unroll
  for (int kk = 0; kk < KSLOT; ++kk) { s[kk] *= 0.0625f; mx = fmaxf(mx, s[kk]); }  // /sqrt(256)
  float sum = 0.f, w[KSLOT];
#pragma unroll
  for (int kk = 0; kk < KSLOT; ++kk) { w[kk] = __expf(s[kk] - mx); sum += w[kk]; }
  const float inv = 1.f / sum;

  float rv[8] = {};
#pragma unroll
  for (int kk = 0; kk < KSLOT; ++kk) {
    const float a = w[kk] * inv;
    const float* vr = vf + kk * DIM + d0;
#pragma unroll
    for (int j = 0; j < 8; ++j) rv[j] = fmaf(a, vr[j], rv[j]);
  }

  // LayerNorm over the full 2048 dims of this token
  float lsum = 0.f, lsq = 0.f;
#pragma unroll
  for (int j = 0; j < 8; ++j) { lsum += rv[j]; lsq += rv[j] * rv[j]; }
  lsum = wave_sum(lsum);
  lsq  = wave_sum(lsq);
  __shared__ float ssum[8], ssq[8];
  __shared__ float smu, srstd;
  if (lane == 0) { ssum[wave] = lsum; ssq[wave] = lsq; }
  __syncthreads();
  if (tid == 0) {
    float ts = 0.f, tq = 0.f;
    for (int i = 0; i < 8; ++i) { ts += ssum[i]; tq += ssq[i]; }
    const float mu = ts / (float)DIM;
    smu = mu;
    srstd = rsqrtf(tq / (float)DIM - mu * mu + LN_EPS);
  }
  __syncthreads();
  const float mu = smu, rstd = srstd;
#pragma unroll
  for (int j = 0; j < 8; ++j) {
    const int d = d0 + j;
    lnOut[(size_t)n * DIM + d] = (bf16)((rv[j] - mu) * rstd * gamma[d] + beta[d]);
  }
}

// ---------------- write scores: [8,N] = wq @ wk.T / sqrt(2048), one token per wave ----------------
__global__ __launch_bounds__(256) void write_scores_k(
    const bf16* __restrict__ wkb, const float* __restrict__ wqf, float* __restrict__ scores) {
  const int wave = threadIdx.x >> 5, lane = threadIdx.x & 31;
  const int n = blockIdx.x * 8 + wave;
  float s[KSLOT] = {};
  for (int t = lane; t < DIM; t += 32) {
    const float x = (float)wkb[(size_t)n * DIM + t];
#pragma unroll
    for (int kk = 0; kk < KSLOT; ++kk) s[kk] = fmaf(x, wqf[kk * DIM + t], s[kk]);
  }
#pragma unroll
  for (int kk = 0; kk < KSLOT; ++kk) s[kk] = wave_sum(s[kk]);
  if (lane == 0) {
#pragma unroll
    for (int kk = 0; kk < KSLOT; ++kk)
      scores[(size_t)kk * N_TOK + n] = s[kk] * 0.022097086912079608f;  // 1/sqrt(2048)
  }
}

// ---------------- row softmax over N=16384 (8 rows) ----------------
__global__ __launch_bounds__(256) void softmax_rows_k(const float* __restrict__ scores,
                                                      float* __restrict__ wattn) {
  const int row = blockIdx.x, tid = threadIdx.x, lane = tid & 31, wave = tid >> 5;
  const float* src = scores + (size_t)row * N_TOK;
  __shared__ float sred[8];
  __shared__ float bmax, bsum;

  float m = -3.402823466e38f;
  for (int i = tid; i < N_TOK; i += 256) m = fmaxf(m, src[i]);
  m = wave_max(m);
  if (lane == 0) sred[wave] = m;
  __syncthreads();
  if (tid == 0) {
    float t = sred[0];
    for (int i = 1; i < 8; ++i) t = fmaxf(t, sred[i]);
    bmax = t;
  }
  __syncthreads();
  m = bmax;

  float s = 0.f;
  for (int i = tid; i < N_TOK; i += 256) s += __expf(src[i] - m);
  s = wave_sum(s);
  __syncthreads();
  if (lane == 0) sred[wave] = s;
  __syncthreads();
  if (tid == 0) {
    float t = 0.f;
    for (int i = 0; i < 8; ++i) t += sred[i];
    bsum = t;
  }
  __syncthreads();
  const float inv = 1.f / bsum;
  for (int i = tid; i < N_TOK; i += 256)
    wattn[(size_t)row * N_TOK + i] = __expf(src[i] - m) * inv;
}

// ---------------- new_info partial sums over 256-token chunks ----------------
__global__ __launch_bounds__(256) void new_info_partial_k(
    const float* __restrict__ wattn, const bf16* __restrict__ wvb, float* __restrict__ partial) {
  const int chunk = blockIdx.y;                       // 64 chunks of 256 tokens
  const int d = blockIdx.x * 256 + threadIdx.x;       // gridDim.x = 8
  __shared__ float aw[KSLOT][256];
#pragma unroll
  for (int kk = 0; kk < KSLOT; ++kk)
    aw[kk][threadIdx.x] = wattn[(size_t)kk * N_TOK + chunk * 256 + threadIdx.x];
  __syncthreads();
  float acc[KSLOT] = {};
  for (int t = 0; t < 256; ++t) {
    const float x = (float)wvb[(size_t)(chunk * 256 + t) * DIM + d];
#pragma unroll
    for (int kk = 0; kk < KSLOT; ++kk) acc[kk] = fmaf(aw[kk][t], x, acc[kk]);
  }
#pragma unroll
  for (int kk = 0; kk < KSLOT; ++kk)
    partial[((size_t)chunk * KSLOT + kk) * DIM + d] = acc[kk];
}

// ---------------- reduce partials + gated EMA memory update ----------------
__global__ void new_memory_k(const float* __restrict__ partial, const float* __restrict__ mem,
                             const float* __restrict__ wgate, float* __restrict__ out) {
  const int idx = blockIdx.x * 256 + threadIdx.x;
  if (idx >= KSLOT * DIM) return;
  float s = 0.f;
  for (int p = 0; p < 64; ++p) s += partial[(size_t)p * (KSLOT * DIM) + idx];
  const float wg = 1.f / (1.f + __expf(-wgate[0]));
  out[idx] = (1.f - wg) * mem[idx] + wg * s;
}

extern "C" void kernel_launch(void* const* d_in, const int* in_sizes, int n_in,
                              void* d_out, int out_size, void* d_ws, size_t ws_size,
                              hipStream_t stream) {
  (void)in_sizes; (void)n_in; (void)out_size; (void)ws_size;
  const float* hidden     = (const float*)d_in[0];
  const float* memory     = (const float*)d_in[1];
  const float* w_read_q   = (const float*)d_in[2];
  const float* w_read_k   = (const float*)d_in[3];
  const float* w_read_v   = (const float*)d_in[4];
  const float* w_read_out = (const float*)d_in[5];
  const float* w_write_q  = (const float*)d_in[6];
  const float* w_write_k  = (const float*)d_in[7];
  const float* w_write_v  = (const float*)d_in[8];
  const float* ln_gamma   = (const float*)d_in[9];
  const float* ln_beta    = (const float*)d_in[10];
  const float* gate       = (const float*)d_in[11];
  const float* wgate      = (const float*)d_in[12];
  float* out = (float*)d_out;

  char* p = (char*)d_ws;
  auto alloc = [&](size_t bytes) { void* r = (void*)p; p += (bytes + 255) & ~(size_t)255; return r; };
  bf16* hidB = (bf16*)alloc((size_t)N_TOK * DIM * 2);
  bf16* wrqB = (bf16*)alloc((size_t)DIM * DIM * 2);
  bf16* wroB = (bf16*)alloc((size_t)DIM * DIM * 2);
  bf16* wwkB = (bf16*)alloc((size_t)DIM * DIM * 2);
  bf16* wwvB = (bf16*)alloc((size_t)DIM * DIM * 2);
  bf16* qB   = (bf16*)alloc((size_t)N_TOK * DIM * 2);
  bf16* lnB  = (bf16*)alloc((size_t)N_TOK * DIM * 2);
  bf16* wkB  = (bf16*)alloc((size_t)N_TOK * DIM * 2);
  bf16* wvB  = (bf16*)alloc((size_t)N_TOK * DIM * 2);
  float* kf      = (float*)alloc((size_t)KSLOT * DIM * 4);
  float* vf      = (float*)alloc((size_t)KSLOT * DIM * 4);
  float* wqf     = (float*)alloc((size_t)KSLOT * DIM * 4);
  float* scores  = (float*)alloc((size_t)KSLOT * N_TOK * 4);
  float* wattn   = (float*)alloc((size_t)KSLOT * N_TOK * 4);
  float* partial = (float*)alloc((size_t)64 * KSLOT * DIM * 4);

  const dim3 blk(256);
  cvt_f32_bf16<<<4096, blk, 0, stream>>>(hidden,     hidB, N_TOK * DIM);
  cvt_f32_bf16<<<1024, blk, 0, stream>>>(w_read_q,   wrqB, DIM * DIM);
  cvt_f32_bf16<<<1024, blk, 0, stream>>>(w_read_out, wroB, DIM * DIM);
  cvt_f32_bf16<<<1024, blk, 0, stream>>>(w_write_k,  wwkB, DIM * DIM);
  cvt_f32_bf16<<<1024, blk, 0, stream>>>(w_write_v,  wwvB, DIM * DIM);
  small_rowmats<<<24, blk, 0, stream>>>(memory, w_read_k, w_read_v, w_write_q, kf, vf, wqf);

  const dim3 gg(DIM / 128, N_TOK / 128);  // (16, 128)
  gemm_bf16_nt<<<gg, blk, 0, stream>>>(hidB, wrqB, nullptr, qB,  nullptr, nullptr, 1);
  gemm_bf16_nt<<<gg, blk, 0, stream>>>(hidB, wwkB, nullptr, wkB, nullptr, nullptr, 1);
  gemm_bf16_nt<<<gg, blk, 0, stream>>>(hidB, wwvB, nullptr, wvB, nullptr, nullptr, 1);

  read_attn_ln<<<N_TOK, blk, 0, stream>>>(qB, kf, vf, ln_gamma, ln_beta, lnB);

  // result = hidden + sigmoid(gate) * (ln @ w_read_out.T), fused epilogue -> d_out
  gemm_bf16_nt<<<gg, blk, 0, stream>>>(lnB, wroB, out, nullptr, hidden, gate, 2);

  write_scores_k<<<N_TOK / 8, blk, 0, stream>>>(wkB, wqf, scores);
  softmax_rows_k<<<KSLOT, blk, 0, stream>>>(scores, wattn);
  new_info_partial_k<<<dim3(DIM / 256, 64), blk, 0, stream>>>(wattn, wvB, partial);
  new_memory_k<<<(KSLOT * DIM) / 256, blk, 0, stream>>>(partial, memory, wgate,
                                                        out + (size_t)N_TOK * DIM);
}
